// ConditionalDetrAttention_49228915146952
// MI455X (gfx1250) — matrix operand order repaired and checked by reading, not compiled, not run on hardware
//
#include <hip/hip_runtime.h>
#include <hip/hip_bf16.h>

typedef __attribute__((ext_vector_type(16))) _Float16 v16h;
typedef __attribute__((ext_vector_type(8)))  float    v8f;

#define EMBED 512
#define OUTD  256
#define NH    8
#define HD    64
#define VHD   32
#define BB    32
#define TT    300
#define SS    850
#define T_TILES 19            // ceil(300/16)
#define M_TILES 600           // (32*300)/16 exactly
#define WPB   4               // waves per block

__device__ __forceinline__ float red_max16(float v) {
  v = fmaxf(v, __shfl_xor(v, 1, 32));
  v = fmaxf(v, __shfl_xor(v, 2, 32));
  v = fmaxf(v, __shfl_xor(v, 4, 32));
  v = fmaxf(v, __shfl_xor(v, 8, 32));
  return v;
}
__device__ __forceinline__ float red_sum16(float v) {
  v += __shfl_xor(v, 1, 32);
  v += __shfl_xor(v, 2, 32);
  v += __shfl_xor(v, 4, 32);
  v += __shfl_xor(v, 8, 32);
  return v;
}

// ---------------------------------------------------------------------------
// Fused flash-attention: one wave = one 16-row Q tile of one (b, h).
// Online softmax over S in 32-key chunks; 4 QK wmmas + 2 PV wmmas per chunk.
// Output written f16 row-major [B*T][256] into workspace.
// ---------------------------------------------------------------------------
__global__ __launch_bounds__(WPB * 32)
void attn_kernel(const float* __restrict__ hs, const float* __restrict__ key,
                 const float* __restrict__ val, const float* __restrict__ mask,
                 _Float16* __restrict__ ws)
{
  __shared__ _Float16 lds_p[WPB][16 * 32];   // P transpose staging, 1KB/wave
  const int wid  = threadIdx.x >> 5;
  const int lane = threadIdx.x & 31;
  const int g    = lane >> 4;                // 16-lane half
  const int ln   = lane & 15;
  const int tile = blockIdx.x * WPB + wid;
  if (tile >= T_TILES) return;
  const int h  = blockIdx.y;
  const int b  = blockIdx.z;
  const int t0 = tile * 16;

  // ---- Q tile in A layout (two 32-wide K-steps), scaling folded in
  const int tq = min(t0 + ln, TT - 1);
  const float* qrow = hs + ((size_t)b * TT + tq) * EMBED + h * HD;
  v16h aq[2];
  #pragma unroll
  for (int ks = 0; ks < 2; ++ks) {
    const float* p1 = qrow + ks * 32 + g * 8;
    const float* p2 = qrow + ks * 32 + 16 + g * 8;
    #pragma unroll
    for (int j = 0; j < 8; ++j) {
      aq[ks][j]     = (_Float16)(p1[j] * 0.125f);   // 1/sqrt(64)
      aq[ks][8 + j] = (_Float16)(p2[j] * 0.125f);
    }
  }

  float mrun[8], lrun[8];
  #pragma unroll
  for (int r = 0; r < 8; ++r) { mrun[r] = -1e30f; lrun[r] = 0.f; }
  v8f o0 = {}; v8f o1 = {};

  const float* kbase = key  + (size_t)b * SS * EMBED + h * HD;
  const float* vbase = val  + (size_t)b * SS * OUTD  + h * VHD;
  const float* mbase = mask + (size_t)b * TT * SS;       // [T][S], head bcast

  for (int s0 = 0; s0 < SS; s0 += 32) {
    // prefetch next key/value chunk rows (global_prefetch_b8)
    if (s0 + 32 + lane < SS) {
      __builtin_prefetch(kbase + (size_t)(s0 + 32 + lane) * EMBED, 0, 3);
      __builtin_prefetch(vbase + (size_t)(s0 + 32 + lane) * OUTD, 0, 3);
    }

    // ---- scores for two 16-key n-tiles
    v8f sc[2];
    #pragma unroll
    for (int nt = 0; nt < 2; ++nt) {
      const int s = s0 + nt * 16 + ln;                   // this lane's key col
      const float* krow = kbase + (size_t)min(s, SS - 1) * EMBED;
      v8f c = {};
      #pragma unroll
      for (int ks = 0; ks < 2; ++ks) {
        v16h bk;                                         // B layout: k = g*8+i (i<8), 16+g*8+(i-8)
        const float* kp = krow + ks * 32 + g * 8;
        #pragma unroll
        for (int i = 0; i < 16; ++i) bk[i] = (_Float16)kp[i < 8 ? i : i + 8];
        c = __builtin_amdgcn_wmma_f32_16x16x32_f16(false, aq[ks], false, bk,
                                                   (short)0, c, false, false);
      }
      // additive mask + kill out-of-range keys
      #pragma unroll
      for (int r = 0; r < 8; ++r) {
        const int t = min(t0 + g * 8 + r, TT - 1);
        if (s < SS) c[r] += mbase[(size_t)t * SS + s];
        else        c[r] = -1e30f;
      }
      sc[nt] = c;
    }

    // ---- online softmax; row m = g*8 + r spans lanes of one 16-lane half
    #pragma unroll
    for (int r = 0; r < 8; ++r) {
      const float tmax  = red_max16(fmaxf(sc[0][r], sc[1][r]));
      const float mnew  = fmaxf(mrun[r], tmax);
      const float scale = __expf(mrun[r] - mnew);
      const float p0 = __expf(sc[0][r] - mnew);
      const float p1 = __expf(sc[1][r] - mnew);
      lrun[r] = lrun[r] * scale + red_sum16(p0 + p1);
      mrun[r] = mnew;
      o0[r] *= scale;
      o1[r] *= scale;
      const int m = g * 8 + r;
      lds_p[wid][m * 32 + ln]      = (_Float16)p0;       // D layout -> LDS
      lds_p[wid][m * 32 + 16 + ln] = (_Float16)p1;
    }
    __builtin_amdgcn_wave_barrier();

    // ---- P back in A layout (two contiguous 8-half ds reads per lane)
    v16h ap;
    #pragma unroll
    for (int j = 0; j < 8; ++j) {
      ap[j]     = lds_p[wid][ln * 32 + g * 8 + j];
      ap[8 + j] = lds_p[wid][ln * 32 + 16 + g * 8 + j];
    }
    __builtin_amdgcn_wave_barrier();

    // ---- V in B layout, two dv halves; accumulate O
    #pragma unroll
    for (int nh = 0; nh < 2; ++nh) {
      v16h bv;
      const int dv = nh * 16 + ln;
      #pragma unroll
      for (int i = 0; i < 16; ++i) {
        const int s = min(s0 + g * 8 + (i < 8 ? i : i + 8), SS - 1);      // p==0 for padded keys
        bv[i] = (_Float16)vbase[(size_t)s * OUTD + dv];
      }
      if (nh == 0)
        o0 = __builtin_amdgcn_wmma_f32_16x16x32_f16(false, ap, false, bv,
                                                    (short)0, o0, false, false);
      else
        o1 = __builtin_amdgcn_wmma_f32_16x16x32_f16(false, ap, false, bv,
                                                    (short)0, o1, false, false);
    }
  }

  // ---- normalize rows, store f16 attn output into workspace [B*T][256]
  #pragma unroll
  for (int r = 0; r < 8; ++r) {
    const int t = t0 + g * 8 + r;
    if (t < TT) {
      const float inv = 1.0f / lrun[r];
      _Float16* wrow = ws + ((size_t)b * TT + t) * OUTD + h * VHD;
      wrow[ln]      = (_Float16)(o0[r] * inv);
      wrow[16 + ln] = (_Float16)(o1[r] * inv);
    }
  }
}

// ---------------------------------------------------------------------------
// out_proj: out[m,n] = sum_k x[m,k] * W[n,k] + bias[n]
// one wave per 16x16 tile, K=256 in 8 wmma steps (W rows are contiguous in k).
// ---------------------------------------------------------------------------
__global__ __launch_bounds__(WPB * 32)
void proj_kernel(const _Float16* __restrict__ x, const float* __restrict__ W,
                 const float* __restrict__ bias, float* __restrict__ out)
{
  const int wid  = threadIdx.x >> 5;
  const int lane = threadIdx.x & 31;
  const int g = lane >> 4, ln = lane & 15;
  const int mt = blockIdx.x * WPB + wid;
  const int nt = blockIdx.y;
  if (mt >= M_TILES) return;

  const _Float16* xrow = x + (size_t)(mt * 16 + ln) * OUTD;   // A: row = ln
  const float*    wr   = W + (size_t)(nt * 16 + ln) * OUTD;   // B: col = ln
  v8f c = {};
  #pragma unroll
  for (int ks = 0; ks < 8; ++ks) {
    v16h a, bw;
    const _Float16* p1 = xrow + ks * 32 + g * 8;
    const _Float16* p2 = xrow + ks * 32 + 16 + g * 8;
    #pragma unroll
    for (int j = 0; j < 8; ++j) { a[j] = p1[j]; a[8 + j] = p2[j]; }
    const float* wp = wr + ks * 32 + g * 8;
    #pragma unroll
    for (int i = 0; i < 16; ++i) bw[i] = (_Float16)wp[i < 8 ? i : i + 8];
    c = __builtin_amdgcn_wmma_f32_16x16x32_f16(false, a, false, bw,
                                               (short)0, c, false, false);
  }
  const float bn = bias[nt * 16 + ln];
  #pragma unroll
  for (int r = 0; r < 8; ++r)
    out[(size_t)(mt * 16 + g * 8 + r) * OUTD + nt * 16 + ln] = c[r] + bn;
}

extern "C" void kernel_launch(void* const* d_in, const int* in_sizes, int n_in,
                              void* d_out, int out_size, void* d_ws, size_t ws_size,
                              hipStream_t stream) {
  const float* hs   = (const float*)d_in[0];   // [B,T,512]
  const float* key  = (const float*)d_in[1];   // [B,S,512]
  const float* val  = (const float*)d_in[2];   // [B,S,256]
  const float* mask = (const float*)d_in[3];   // [B,1,T,S]
  const float* W    = (const float*)d_in[4];   // [256,256]
  const float* bias = (const float*)d_in[5];   // [256]
  float* out = (float*)d_out;                  // [B,T,256] f32
  _Float16* ws = (_Float16*)d_ws;              // [B*T,256] f16 staging (4.9 MB)

  dim3 g1((T_TILES + WPB - 1) / WPB, NH, BB);  // 5 x 8 x 32 blocks of 128
  attn_kernel<<<g1, WPB * 32, 0, stream>>>(hs, key, val, mask, ws);

  dim3 g2(M_TILES / WPB, OUTD / 16);           // 150 x 16 blocks of 128
  proj_kernel<<<g2, WPB * 32, 0, stream>>>(ws, W, bias, out);
}
